// CaseGNN_86423331930378
// MI455X (gfx1250) — matrix operand order repaired and checked by reading, not compiled, not run on hardware
//
#include <hip/hip_runtime.h>
#include <hip/hip_bf16.h>
#include <math.h>

typedef __bf16 bf16;
typedef __attribute__((ext_vector_type(16))) __bf16 v16bf;
typedef __attribute__((ext_vector_type(8)))  float   v8f;

#define D_DIM 768
#define NEG_SLOPE 0.2f

#define MT 128
#define NT 64
#define KT 32
#define LPAD 40   // padded LDS row (32 data + 8 pad bf16) -> 80B, 16B aligned

__device__ __forceinline__ bf16 f2bf(float f) { return (bf16)f; }

// ---- CDNA5 async global->LDS copy (memory -> LDS, no VGPR data, ASYNCcnt) ----
// vdst = per-lane LDS byte offset, vaddr = per-lane 64-bit global address.
__device__ __forceinline__ void async_cp_b128(void* lds, const void* g) {
  unsigned lds_off = (unsigned)(unsigned long long)(uintptr_t)lds;  // low 32b of flat = LDS offset
  asm volatile("global_load_async_to_lds_b128 %0, %1, off"
               :: "v"(lds_off), "v"(g) : "memory");
}
__device__ __forceinline__ void wait_async0() {
#if __has_builtin(__builtin_amdgcn_s_wait_asynccnt)
  __builtin_amdgcn_s_wait_asynccnt(0);
#else
  asm volatile("s_wait_asynccnt 0" ::: "memory");
#endif
}

// ---------------- convert f32 -> bf16 ----------------
__global__ void k_cvt_bf16(const float* __restrict__ in, bf16* __restrict__ out, long n) {
  long i = (long)blockIdx.x * blockDim.x + threadIdx.x;
  if (i < n) out[i] = f2bf(in[i]);
}

// ---------------- bf16 WMMA GEMM: C[M,N] = A[M,K] * B[K,N] ----------------
__global__ __launch_bounds__(256) void k_gemm_bf16(
    const bf16* __restrict__ A, const bf16* __restrict__ B,
    float* __restrict__ C, int M, int N, int K)
{
  __shared__ bf16 As[2][MT][LPAD];       // row-major A tile (async-staged)
  __shared__ bf16 Bs[2][NT][LPAD];       // transposed B tile: Bs[col][k]

  const int t    = threadIdx.x;
  const int wave = t >> 5;
  const int lane = t & 31;
  const int half = lane >> 4;            // 0: lanes 0-15, 1: lanes 16-31
  const int lrow = lane & 15;

  const int m0   = blockIdx.y * MT;
  const int n0   = blockIdx.x * NT;
  const int wrow = wave * 16;

  // cooperative staging indices
  const int ar = t >> 1;                 // 0..127 (A tile row)
  const int ac = (t & 1) * 16;           // 0 or 16 (A tile col base)
  const int bk = t >> 3;                 // 0..31  (B tile k row)
  const int bc = (t & 7) * 8;            // 0..56  (B tile col base)

  v8f zero = {};
  v8f acc[4];
  #pragma unroll
  for (int i = 0; i < 4; ++i) acc[i] = zero;

  int aclamp = m0 + ar; if (aclamp >= M) aclamp = M - 1;
  const bf16* arow_g = A + (size_t)aclamp * K + ac;

  auto stage = [&](int buf, int k0) {
    // A tile: two async 16B memory->LDS transfers per lane (no VGPR bounce)
    async_cp_b128(&As[buf][ar][ac],     arow_g + k0);
    async_cp_b128(&As[buf][ar][ac + 8], arow_g + k0 + 8);
    // B tile: load 8 contiguous bf16 of one k-row, scatter transposed into LDS
    uint4 braw = *(const uint4*)(B + (size_t)(k0 + bk) * N + (n0 + bc));
    const bf16* pb = (const bf16*)&braw;
    #pragma unroll
    for (int i = 0; i < 8; ++i) Bs[buf][bc + i][bk] = pb[i];
  };

  const int ksteps = K / KT;
  stage(0, 0);
  int buf = 0;

  for (int kt = 0; kt < ksteps; ++kt) {
    wait_async0();        // my wave's async LDS writes are complete
    __syncthreads();      // everyone's staging (async A + ds B) visible
    if (kt + 1 < ksteps) {
      if (kt + 2 < ksteps)   // L2 prefetch of B tile-after-next (global_prefetch_b8)
        __builtin_prefetch(B + (size_t)((kt + 2) * KT + bk) * N + (n0 + bc), 0, 1);
      stage(buf ^ 1, (kt + 1) * KT);
    }

    // Preload A fragment + all 4 B fragments, then chain the 4 WMMAs back-to-back.
    union U { uint4 u[2]; v16bf v; };
    U af, bfr[4];
    // A: 16x32 bf16 per ISA layout (lanes 0-15: K 0..7 | 16..23; lanes 16-31: K 8..15 | 24..31)
    const bf16* arow = &As[buf][wrow + lrow][half * 8];
    af.u[0] = *(const uint4*)(arow);
    af.u[1] = *(const uint4*)(arow + 16);
    #pragma unroll
    for (int ci = 0; ci < 4; ++ci) {
      // B: col = 16*ci + lrow, same K layout as A (lanes 0-15: K 0..7 | 16..23; lanes 16-31: K 8..15 | 24..31)
      const bf16* bcol = &Bs[buf][ci * 16 + lrow][half * 8];
      bfr[ci].u[0] = *(const uint4*)(bcol);
      bfr[ci].u[1] = *(const uint4*)(bcol + 16);
    }
    #pragma unroll
    for (int ci = 0; ci < 4; ++ci)
      acc[ci] = __builtin_amdgcn_wmma_f32_16x16x32_bf16(
          false, af.v, false, bfr[ci].v, (short)0, acc[ci], false, false);

    buf ^= 1;
  }

  // C/D layout: VGPR r -> M = half*8 + r, N = lrow (per 16-col subtile)
  #pragma unroll
  for (int ci = 0; ci < 4; ++ci) {
    #pragma unroll
    for (int r = 0; r < 8; ++r) {
      int grow = m0 + wrow + half * 8 + r;
      int gcol = n0 + ci * 16 + lrow;
      if (grow < M) C[(size_t)grow * N + gcol] = acc[ci][r];
    }
  }
}

// ---------------- wave-per-row dual dot: el/er ----------------
__global__ __launch_bounds__(256) void k_rowdot2(const float* __restrict__ X,
    const float* __restrict__ a1, const float* __restrict__ a2,
    float* __restrict__ o1, float* __restrict__ o2, int rows)
{
  int row  = blockIdx.x * 8 + (threadIdx.x >> 5);
  int lane = threadIdx.x & 31;
  if (row >= rows) return;
  const float* x = X + (size_t)row * D_DIM;
  float s1 = 0.f, s2 = 0.f;
  for (int d = lane; d < D_DIM; d += 32) {
    float v = x[d];
    s1 += v * a1[d];
    s2 += v * a2[d];
  }
  #pragma unroll
  for (int off = 16; off > 0; off >>= 1) {
    s1 += __shfl_xor(s1, off, 32);
    s2 += __shfl_xor(s2, off, 32);
  }
  if (lane == 0) { o1[row] = s1; o2[row] = s2; }
}

__global__ __launch_bounds__(256) void k_rowdot1(const float* __restrict__ X,
    const float* __restrict__ a1, float* __restrict__ o1, int rows)
{
  int row  = blockIdx.x * 8 + (threadIdx.x >> 5);
  int lane = threadIdx.x & 31;
  if (row >= rows) return;
  const float* x = X + (size_t)row * D_DIM;
  float s1 = 0.f;
  for (int d = lane; d < D_DIM; d += 32) s1 += x[d] * a1[d];
  #pragma unroll
  for (int off = 16; off > 0; off >>= 1) s1 += __shfl_xor(s1, off, 32);
  if (lane == 0) o1[row] = s1;
}

// ---------------- edge score: leaky_relu(el[src]+er[dst]+ee) ----------------
__global__ void k_edge_score(const float* __restrict__ el, const float* __restrict__ er,
                             const float* __restrict__ ee, const int* __restrict__ src,
                             const int* __restrict__ dst, float* __restrict__ score, int E)
{
  int e = blockIdx.x * blockDim.x + threadIdx.x;
  if (e >= E) return;
  float v = el[src[e]] + er[dst[e]] + ee[e];
  score[e] = v > 0.f ? v : NEG_SLOPE * v;
}

__global__ void k_seg_init(float* __restrict__ smax, float* __restrict__ ssum, int N) {
  int i = blockIdx.x * blockDim.x + threadIdx.x;
  if (i < N) { smax[i] = -INFINITY; ssum[i] = 0.f; }
}

__device__ void atomicMaxF(float* addr, float val) {
  unsigned* ua = (unsigned*)addr;
  unsigned old = *ua;
  while (__uint_as_float(old) < val) {
    unsigned assumed = old;
    old = atomicCAS(ua, assumed, __float_as_uint(val));
    if (old == assumed) break;
  }
}

__global__ void k_seg_max(const float* __restrict__ score, const int* __restrict__ dst,
                          float* __restrict__ smax, int E)
{
  int e = blockIdx.x * blockDim.x + threadIdx.x;
  if (e < E) atomicMaxF(&smax[dst[e]], score[e]);
}

__global__ void k_edge_exp(const float* __restrict__ score, const int* __restrict__ dst,
                           const float* __restrict__ smax, float* __restrict__ w,
                           float* __restrict__ ssum, int E)
{
  int e = blockIdx.x * blockDim.x + threadIdx.x;
  if (e >= E) return;
  int d = dst[e];
  float m = smax[d];
  if (!(m >= -3.0e38f && m <= 3.0e38f)) m = 0.f;   // non-finite -> 0 (ref semantics)
  float v = __expf(score[e] - m);
  w[e] = v;
  atomicAdd(&ssum[d], v);
}

__global__ void k_edge_norm(float* __restrict__ w, const int* __restrict__ dst,
                            const float* __restrict__ ssum, int E)
{
  int e = blockIdx.x * blockDim.x + threadIdx.x;
  if (e >= E) return;
  float s = ssum[dst[e]];
  w[e] = w[e] / (s > 0.f ? s : 1.f);
}

// ---------------- rst = x + bias (residual + bias init) ----------------
__global__ void k_rst_init(const float* __restrict__ x, const float* __restrict__ bias,
                           float* __restrict__ rst, long nd)
{
  long i = (long)blockIdx.x * blockDim.x + threadIdx.x;
  if (i >= nd) return;
  int d = (int)(i % D_DIM);
  rst[i] = x[i] + bias[d];
}

// ---------------- message scatter: rst[dst] += (fs[src]+fe[e]) * a[e] ----------------
__global__ void k_scatter(const float* __restrict__ fs, const float* __restrict__ fe,
                          const float* __restrict__ a, const int* __restrict__ src,
                          const int* __restrict__ dst, float* __restrict__ rst, int E)
{
  const int CH = D_DIM / 4;  // 192 float4 chunks per edge
  long t = (long)blockIdx.x * blockDim.x + threadIdx.x;
  long e = t / CH;
  if (e >= E) return;
  int d4 = (int)(t % CH) * 4;
  float av = a[e];
  int s  = src[e];
  int dd = dst[e];
  float4 vs = *(const float4*)(fs + (size_t)s * D_DIM + d4);
  float4 ve = *(const float4*)(fe + (size_t)e * D_DIM + d4);
  float* o = rst + (size_t)dd * D_DIM + d4;
  atomicAdd(o + 0, (vs.x + ve.x) * av);
  atomicAdd(o + 1, (vs.y + ve.y) * av);
  atomicAdd(o + 2, (vs.z + ve.z) * av);
  atomicAdd(o + 3, (vs.w + ve.w) * av);
}

// ---------------- fused ReLU (in place) + bf16 reconvert for next layer ----------------
__global__ void k_relu_cvt(float* __restrict__ h, bf16* __restrict__ hb, long n) {
  long i = (long)blockIdx.x * blockDim.x + threadIdx.x;
  if (i >= n) return;
  float v = h[i];
  v = v > 0.f ? v : 0.f;
  h[i]  = v;
  hb[i] = f2bf(v);
}

// ---------------- final gather of graph_offsets rows ----------------
__global__ void k_gather(const float* __restrict__ rst, const int* __restrict__ offs,
                         float* __restrict__ out, int B)
{
  int i = blockIdx.x * blockDim.x + threadIdx.x;
  if (i >= B * D_DIM) return;
  int b = i / D_DIM;
  int d = i % D_DIM;
  out[i] = rst[(size_t)offs[b] * D_DIM + d];
}

// ---------------- one EdgeGAT layer ----------------
static void edge_gat_layer(const bf16* Xb, const bf16* Eb, const bf16* Wb, const bf16* Web,
                           const float* al, const float* ar, const float* ae,
                           const float* bias, const float* xres,
                           const int* src, const int* dst,
                           float* fs, float* fe, float* el, float* er, float* ee,
                           float* score, float* smax, float* ssum, float* acoef,
                           float* rst, int N, int E, hipStream_t stream)
{
  dim3 blk(256);
  dim3 gN(D_DIM / NT, (N + MT - 1) / MT);
  dim3 gE(D_DIM / NT, (E + MT - 1) / MT);
  k_gemm_bf16<<<gN, blk, 0, stream>>>(Xb, Wb, fs, N, D_DIM, D_DIM);
  k_gemm_bf16<<<gE, blk, 0, stream>>>(Eb, Web, fe, E, D_DIM, D_DIM);
  k_rowdot2<<<(N + 7) / 8, blk, 0, stream>>>(fs, al, ar, el, er, N);
  k_rowdot1<<<(E + 7) / 8, blk, 0, stream>>>(fe, ae, ee, E);
  k_edge_score<<<(E + 255) / 256, blk, 0, stream>>>(el, er, ee, src, dst, score, E);
  k_seg_init<<<(N + 255) / 256, blk, 0, stream>>>(smax, ssum, N);
  k_seg_max<<<(E + 255) / 256, blk, 0, stream>>>(score, dst, smax, E);
  k_edge_exp<<<(E + 255) / 256, blk, 0, stream>>>(score, dst, smax, acoef, ssum, E);
  k_edge_norm<<<(E + 255) / 256, blk, 0, stream>>>(acoef, dst, ssum, E);
  long nd = (long)N * D_DIM;
  k_rst_init<<<(int)((nd + 255) / 256), blk, 0, stream>>>(xres, bias, rst, nd);
  long sc = (long)E * (D_DIM / 4);
  k_scatter<<<(int)((sc + 255) / 256), blk, 0, stream>>>(fs, fe, acoef, src, dst, rst, E);
}

extern "C" void kernel_launch(void* const* d_in, const int* in_sizes, int n_in,
                              void* d_out, int out_size, void* d_ws, size_t ws_size,
                              hipStream_t stream) {
  const float* node  = (const float*)d_in[0];
  const float* edgef = (const float*)d_in[1];
  const int*   src   = (const int*)d_in[2];
  const int*   dst   = (const int*)d_in[3];
  const int*   goff  = (const int*)d_in[4];
  const float* W1    = (const float*)d_in[5];
  const float* We1   = (const float*)d_in[6];
  const float* al1   = (const float*)d_in[7];
  const float* ar1   = (const float*)d_in[8];
  const float* ae1   = (const float*)d_in[9];
  const float* b1    = (const float*)d_in[10];
  const float* W2    = (const float*)d_in[11];
  const float* We2   = (const float*)d_in[12];
  const float* al2   = (const float*)d_in[13];
  const float* ar2   = (const float*)d_in[14];
  const float* ae2   = (const float*)d_in[15];
  const float* b2    = (const float*)d_in[16];

  const int N = in_sizes[0] / D_DIM;
  const int E = in_sizes[1] / D_DIM;
  const int B = in_sizes[4];

  // ---- carve workspace ----
  char* ws = (char*)d_ws;
  size_t off = 0;
  auto carve = [&](size_t bytes) -> void* {
    void* p = ws + off;
    off += (bytes + 255) & ~(size_t)255;
    return p;
  };
  bf16*  Xb   = (bf16*)carve((size_t)N * D_DIM * sizeof(bf16));
  bf16*  Eb   = (bf16*)carve((size_t)E * D_DIM * sizeof(bf16));
  bf16*  W1b  = (bf16*)carve((size_t)D_DIM * D_DIM * sizeof(bf16));
  bf16*  We1b = (bf16*)carve((size_t)D_DIM * D_DIM * sizeof(bf16));
  bf16*  W2b  = (bf16*)carve((size_t)D_DIM * D_DIM * sizeof(bf16));
  bf16*  We2b = (bf16*)carve((size_t)D_DIM * D_DIM * sizeof(bf16));
  float* fs   = (float*)carve((size_t)N * D_DIM * sizeof(float));
  float* fe   = (float*)carve((size_t)E * D_DIM * sizeof(float));
  float* rst1 = (float*)carve((size_t)N * D_DIM * sizeof(float));
  float* rst2 = (float*)carve((size_t)N * D_DIM * sizeof(float));
  float* el   = (float*)carve((size_t)N * sizeof(float));
  float* er   = (float*)carve((size_t)N * sizeof(float));
  float* ee   = (float*)carve((size_t)E * sizeof(float));
  float* scr  = (float*)carve((size_t)E * sizeof(float));
  float* smax = (float*)carve((size_t)N * sizeof(float));
  float* ssum = (float*)carve((size_t)N * sizeof(float));
  float* acoef= (float*)carve((size_t)E * sizeof(float));

  dim3 blk(256);
  long nd = (long)N * D_DIM;
  long ed = (long)E * D_DIM;
  long dd = (long)D_DIM * D_DIM;

  // ---- one-time bf16 conversions ----
  k_cvt_bf16<<<(int)((nd + 255) / 256), blk, 0, stream>>>(node,  Xb,   nd);
  k_cvt_bf16<<<(int)((ed + 255) / 256), blk, 0, stream>>>(edgef, Eb,   ed);
  k_cvt_bf16<<<(int)((dd + 255) / 256), blk, 0, stream>>>(W1,    W1b,  dd);
  k_cvt_bf16<<<(int)((dd + 255) / 256), blk, 0, stream>>>(We1,   We1b, dd);
  k_cvt_bf16<<<(int)((dd + 255) / 256), blk, 0, stream>>>(W2,    W2b,  dd);
  k_cvt_bf16<<<(int)((dd + 255) / 256), blk, 0, stream>>>(We2,   We2b, dd);

  // ---- layer 1 ----
  edge_gat_layer(Xb, Eb, W1b, We1b, al1, ar1, ae1, b1, node, src, dst,
                 fs, fe, el, er, ee, scr, smax, ssum, acoef, rst1, N, E, stream);

  // ---- ReLU + reconvert h to bf16 (Xb reused) ----
  k_relu_cvt<<<(int)((nd + 255) / 256), blk, 0, stream>>>(rst1, Xb, nd);

  // ---- layer 2 (fs/fe reused) ----
  edge_gat_layer(Xb, Eb, W2b, We2b, al2, ar2, ae2, b2, rst1, src, dst,
                 fs, fe, el, er, ee, scr, smax, ssum, acoef, rst2, N, E, stream);

  // ---- gather first-node embeddings ----
  k_gather<<<(B * D_DIM + 255) / 256, blk, 0, stream>>>(rst2, goff, (float*)d_out, B);
}
